// MultiheadSelfAttention_36593121362277
// MI455X (gfx1250) — matrix operand order repaired and checked by reading, not compiled, not run on hardware
//
#include <hip/hip_runtime.h>
#include <hip/hip_bf16.h>

#define DEV __device__ __forceinline__

typedef __attribute__((ext_vector_type(16))) __bf16 v16bf;
typedef __attribute__((ext_vector_type(8)))  __bf16 v8bf;
typedef __attribute__((ext_vector_type(8)))  float  v8f;
typedef __attribute__((ext_vector_type(4)))  unsigned int v4u;
typedef __attribute__((ext_vector_type(8)))  int    v8i;
typedef __attribute__((ext_vector_type(4)))  int    v4i;

static constexpr int Bc  = 4;
static constexpr int Sc  = 2048;
static constexpr int Dc  = 2048;
static constexpr int Hc  = 16;
static constexpr int DKc = 128;
static constexpr int Mrows = Bc * Sc;   // 8192

DEV int lane_id() { return (int)(threadIdx.x & 31u); }

// A-fragment (16xK=32, bf16), row = lane&15, K interleave per ISA:
// elems 0..7 -> K = hi*8+i ; elems 8..15 -> K = 16+hi*8+i  (hi = lane>>4)
DEV v16bf load_a_frag(const __bf16* base, int ld) {
  const int lane = lane_id();
  const __bf16* p = base + (size_t)(lane & 15) * ld + ((lane >> 4) << 3);
  v8bf lo = *(const v8bf*)(p);
  v8bf hi = *(const v8bf*)(p + 16);
  return __builtin_shufflevector(lo, hi, 0,1,2,3,4,5,6,7,8,9,10,11,12,13,14,15);
}

// B-fragment (K=32 x 16, bf16), column = lane&15, K interleave per ISA:
// elems 0..7 -> K = hi*8+i ; elems 8..15 -> K = 16+hi*8+i  (hi = lane>>4)
DEV v16bf load_b_frag(const __bf16* base, int ld) {
  const int lane = lane_id();
  const __bf16* p = base + (size_t)(lane & 15) * ld + ((lane >> 4) << 3);
  v8bf lo = *(const v8bf*)(p);
  v8bf hi = *(const v8bf*)(p + 16);
  return __builtin_shufflevector(lo, hi, 0,1,2,3,4,5,6,7,8,9,10,11,12,13,14,15);
}

DEV v8f wmma_bf16(v16bf a, v16bf b, v8f c) {
  return __builtin_amdgcn_wmma_f32_16x16x32_bf16(
      /*neg_a=*/false, a, /*neg_b=*/false, b,
      /*c_mod=*/(short)0, c, /*reuse_a=*/false, /*reuse_b=*/false);
}

// Row-wise reductions across the 16 lanes of a wave half (C-layout rows).
DEV float rowmax16(float v) {
#pragma unroll
  for (int off = 8; off; off >>= 1) v = fmaxf(v, __shfl_xor(v, off, 16));
  return v;
}
DEV float rowsum16(float v) {
#pragma unroll
  for (int off = 8; off; off >>= 1) v += __shfl_xor(v, off, 16);
  return v;
}

// Flat shared-memory pointer -> 32-bit LDS byte offset (aperture low bits).
DEV unsigned lds_off(const void* p) {
  return (unsigned)(unsigned long long)(uintptr_t)p;
}

// ---------------------------------------------------------------------------
// Tensor Data Mover: 2D tile (tile1 rows x tile0 elems of bf16) from global
// (row stride = stride_elems) into LDS at lds_byte_off, packed row-major
// (ld = tile0). D# bitfields per CDNA5 ISA.
// One DMA per executing wave; tracked with TENSORcnt.
// ---------------------------------------------------------------------------
DEV void tdm_load_2d(const __bf16* gptr, unsigned lds_byte_off,
                     int tile0, int tile1, int stride_elems) {
  const unsigned long long ga = (unsigned long long)(uintptr_t)gptr;
  v4u g0;
  g0[0] = 1u;                                          // count=1, user mode
  g0[1] = lds_byte_off;                                // lds_addr
  g0[2] = (unsigned)(ga & 0xffffffffull);              // global_addr[31:0]
  g0[3] = (unsigned)((ga >> 32) & 0x01ffffffull)       // global_addr[56:32]
        | (2u << 30);                                  // type=2 ("image")
  const unsigned td0 = (unsigned)tile0;                // tensor_dim0 = tile
  const unsigned td1 = (unsigned)tile1;                // tensor_dim1 = tile
  const unsigned st0 = (unsigned)stride_elems;         // tensor_dim0_stride
  v8i g1;
  g1[0] = (int)(1u << 16);                             // data_size=1 -> 2B
  g1[1] = (int)((td0 & 0xffffu) << 16);                // tensor_dim0[15:0]
  g1[2] = (int)(((td0 >> 16) & 0xffffu) | ((td1 & 0xffffu) << 16));
  g1[3] = (int)(((td1 >> 16) & 0xffffu) | ((unsigned)tile0 << 16)); // tile_dim0
  g1[4] = (int)((unsigned)tile1 & 0xffffu);            // tile_dim1, tile_dim2=0
  g1[5] = (int)st0;                                    // dim0_stride[31:0]
  g1[6] = 0;                                           // dim0_stride[47:32], dim1_stride lo
  g1[7] = 0;
  v4i z4 = {0, 0, 0, 0};
#if defined(__clang_major__) && (__clang_major__ >= 23)
  v8i z8 = {0, 0, 0, 0, 0, 0, 0, 0};
  __builtin_amdgcn_tensor_load_to_lds(g0, g1, z4, z4, z8, 0);
#else
  __builtin_amdgcn_tensor_load_to_lds(g0, g1, z4, z4, 0);
#endif
}

// ---------------------------------------------------------------------------
// f32 -> bf16 conversion
// ---------------------------------------------------------------------------
__global__ void f32_to_bf16_kernel(const float* __restrict__ in,
                                   __bf16* __restrict__ out, size_t n) {
  size_t i = (size_t)blockIdx.x * blockDim.x + threadIdx.x;
  if (i < n) out[i] = (__bf16)in[i];
}

// ---------------------------------------------------------------------------
// C = A[M,K] * W[N,K]^T  (bf16 inputs, f32 accumulate via WMMA)
// Block: 256 threads = 8 waves (2x4). Wave tile = 32x64. Block tile = 64x256.
// K-loop double-buffers 64x32 (A) and 256x32 (B) tiles in LDS via TDM;
// wave 0 drives the DMA, s_wait_tensorcnt + barrier hand off to consumers.
// MODE 0: f32 row-major [M,N] output (final projection)
// MODE 1: bf16 per-head  [b,h,s,d]  (Q,K)
// MODE 2: bf16 per-head transposed [b,h,d,s]  (V)
// ---------------------------------------------------------------------------
template <int MODE>
__global__ __launch_bounds__(256)
void gemm_xwt(const __bf16* __restrict__ A, const __bf16* __restrict__ W,
              void* __restrict__ Out, int M, int N, int K) {
  __shared__ alignas(16) __bf16 ldsA[2][64 * 32];    //  8 KB
  __shared__ alignas(16) __bf16 ldsB[2][256 * 32];   // 32 KB
  const int wave = (int)(threadIdx.x >> 5);
  const int rowblk = blockIdx.x * 64;
  const int colblk = blockIdx.y * 256;
  const int row0 = (wave >> 2) * 32;   // local row of wave tile
  const int col0 = (wave & 3) * 64;    // local col of wave tile
  v8f acc[2][4] = {};
  const int ksteps = K / 32;

  if (wave == 0) {                     // prologue: stage 0
    tdm_load_2d(A + (size_t)rowblk * K, lds_off(&ldsA[0][0]), 32, 64, K);
    tdm_load_2d(W + (size_t)colblk * K, lds_off(&ldsB[0][0]), 32, 256, K);
  }
  for (int kt = 0; kt < ksteps; ++kt) {
    const int cur = kt & 1;
    if (wave == 0) {
      if (kt + 1 < ksteps) {           // issue next stage, then wait for cur
        const int k0n = (kt + 1) * 32;
        tdm_load_2d(A + (size_t)rowblk * K + k0n, lds_off(&ldsA[cur ^ 1][0]), 32, 64, K);
        tdm_load_2d(W + (size_t)colblk * K + k0n, lds_off(&ldsB[cur ^ 1][0]), 32, 256, K);
        __builtin_amdgcn_s_wait_tensorcnt(2);   // cur pair done (in-order)
      } else {
        __builtin_amdgcn_s_wait_tensorcnt(0);
      }
    }
    __syncthreads();                   // publish stage `cur` to all waves
    v16bf a0 = load_a_frag(&ldsA[cur][(size_t)row0 * 32], 32);
    v16bf a1 = load_a_frag(&ldsA[cur][(size_t)(row0 + 16) * 32], 32);
#pragma unroll
    for (int t = 0; t < 4; ++t) {
      v16bf b = load_b_frag(&ldsB[cur][(size_t)(col0 + t * 16) * 32], 32);
      acc[0][t] = wmma_bf16(a0, b, acc[0][t]);
      acc[1][t] = wmma_bf16(a1, b, acc[1][t]);
    }
    __syncthreads();                   // all consumed before buffer reuse
  }

  const int lane = lane_id();
  const int n = lane & 15, mh = (lane >> 4) * 8;
#pragma unroll
  for (int i = 0; i < 2; ++i) {
#pragma unroll
    for (int t = 0; t < 4; ++t) {
#pragma unroll
      for (int j = 0; j < 8; ++j) {
        const int r = rowblk + row0 + i * 16 + mh + j;
        const int e = colblk + col0 + t * 16 + n;
        const float v = acc[i][t][j];
        if (MODE == 0) {
          ((float*)Out)[(size_t)r * N + e] = v;
        } else {
          const int b = r >> 11, s = r & (Sc - 1);     // r = b*S + s
          const int h = e >> 7, d = e & (DKc - 1);     // e = h*DK + d
          if (MODE == 1)
            ((__bf16*)Out)[(((size_t)(b * Hc + h) * Sc + s) << 7) + d] = (__bf16)v;
          else
            ((__bf16*)Out)[(((size_t)(b * Hc + h) * DKc + d) << 11) + s] = (__bf16)v;
        }
      }
    }
  }
}

// ---------------------------------------------------------------------------
// RoPE in place on Qh, Kh : [b,h,s,d] bf16, one thread per (b,h,s,pair)
// ---------------------------------------------------------------------------
__global__ void rope_kernel(__bf16* __restrict__ Qh, __bf16* __restrict__ Kh,
                            const int* __restrict__ pos, size_t total) {
  size_t idx = (size_t)blockIdx.x * blockDim.x + threadIdx.x;
  if (idx >= total) return;
  const int i = (int)(idx & 63);        // pair index, d_k/2 = 64
  const size_t row = idx >> 6;          // (b*H + h)*S + s
  const int s = (int)(row & (Sc - 1));
  const size_t base = row * DKc;
  const float p = (float)pos[s];
  // theta^(-2i/d_k) = exp(-(2i/128) * ln(10000))
  const float freq = __expf(-(2.0f * (float)i / 128.0f) * 9.210340371976184f);
  float sn, cs;
  __sincosf(p * freq, &sn, &cs);
  float q1 = (float)Qh[base + 2 * i], q2 = (float)Qh[base + 2 * i + 1];
  Qh[base + 2 * i]     = (__bf16)(q1 * cs - q2 * sn);
  Qh[base + 2 * i + 1] = (__bf16)(q1 * sn + q2 * cs);
  float k1 = (float)Kh[base + 2 * i], k2 = (float)Kh[base + 2 * i + 1];
  Kh[base + 2 * i]     = (__bf16)(k1 * cs - k2 * sn);
  Kh[base + 2 * i + 1] = (__bf16)(k1 * sn + k2 * cs);
}

// ---------------------------------------------------------------------------
// Flash attention: one wave (32 threads) per 16-query-row tile.
// Qh,Kh : [b,h,s,d] bf16 ; Vt : [b,h,d,s] bf16 ; Att : [b,s,h*d] bf16
// ---------------------------------------------------------------------------
__global__ __launch_bounds__(32)
void attn_kernel(const __bf16* __restrict__ Qh, const __bf16* __restrict__ Kh,
                 const __bf16* __restrict__ Vt, __bf16* __restrict__ Att) {
  const int bid = blockIdx.x;           // B*H*(S/16) = 8192
  const int qt = bid & 127;             // S/16 tiles
  const int bh = bid >> 7;              // b*H + h
  const int q0 = qt * 16;
  const __bf16* Qbase = Qh + ((size_t)bh * Sc + q0) * DKc;
  const __bf16* Kbase = Kh + (size_t)bh * Sc * DKc;
  const __bf16* Vbase = Vt + (size_t)bh * DKc * Sc;
  const int lane = lane_id();
  const int n = lane & 15, mh = (lane >> 4) * 8;

  v16bf qf[4];
#pragma unroll
  for (int c = 0; c < 4; ++c) qf[c] = load_a_frag(Qbase + c * 32, DKc);

  v8f o[8] = {};
  float mrow[8], lrow[8];
#pragma unroll
  for (int j = 0; j < 8; ++j) { mrow[j] = -3.0e38f; lrow[j] = 0.0f; }

  __shared__ alignas(16) __bf16 pbuf[16 * 32];
  const float scale = 0.08838834764831845f;   // 1/sqrt(128)

  const int ktiles = (q0 + 16 + 31) >> 5;     // causal bound, 32 keys/tile
  for (int kt = 0; kt < ktiles; ++kt) {
    const int k0 = kt * 32;
    v8f s0 = {}, s1 = {};
#pragma unroll
    for (int c = 0; c < 4; ++c) {             // d_k = 4 x 32
      v16bf kf0 = load_b_frag(Kbase + (size_t)k0 * DKc + c * 32, DKc);
      v16bf kf1 = load_b_frag(Kbase + (size_t)(k0 + 16) * DKc + c * 32, DKc);
      s0 = wmma_bf16(qf[c], kf0, s0);
      s1 = wmma_bf16(qf[c], kf1, s1);
    }
    // scale + causal mask + online softmax update
#pragma unroll
    for (int j = 0; j < 8; ++j) {
      const int qrow = q0 + mh + j;
      float v0 = s0[j] * scale;
      float v1 = s1[j] * scale;
      if (k0 + n > qrow)      v0 = -3.0e38f;
      if (k0 + 16 + n > qrow) v1 = -3.0e38f;
      const float rmax = rowmax16(fmaxf(v0, v1));
      const float mnew = fmaxf(mrow[j], rmax);
      const float alpha = __expf(mrow[j] - mnew);
      const float p0 = __expf(v0 - mnew);
      const float p1 = __expf(v1 - mnew);
      lrow[j] = lrow[j] * alpha + rowsum16(p0 + p1);
      mrow[j] = mnew;
#pragma unroll
      for (int t = 0; t < 8; ++t) o[t][j] *= alpha;
      pbuf[(mh + j) * 32 + n]      = (__bf16)p0;
      pbuf[(mh + j) * 32 + 16 + n] = (__bf16)p1;
    }
    __syncthreads();                     // single-wave WG: just orders LDS
    v16bf pf = load_a_frag(pbuf, 32);    // P as 16x32 A-fragment
    __syncthreads();
#pragma unroll
    for (int t = 0; t < 8; ++t) {        // O(16x128) += P(16x32) * V(32x128)
      v16bf vf = load_b_frag(Vbase + (size_t)(t * 16) * Sc + k0, Sc);
      o[t] = wmma_bf16(pf, vf, o[t]);
    }
  }

  // normalize + store merged [b, s, h*DK + d] as bf16 for the final GEMM
  const int b = bh >> 4, h = bh & 15;
#pragma unroll
  for (int t = 0; t < 8; ++t) {
    const float invl_unused = 0.0f; (void)invl_unused;
#pragma unroll
    for (int j = 0; j < 8; ++j) {
      const float val = o[t][j] / lrow[j];
      const int srow = q0 + mh + j;
      const size_t idx = ((size_t)(b * Sc + srow) * Dc) + h * DKc + t * 16 + n;
      Att[idx] = (__bf16)val;
    }
  }
}

// ---------------------------------------------------------------------------
extern "C" void kernel_launch(void* const* d_in, const int* in_sizes, int n_in,
                              void* d_out, int out_size, void* d_ws, size_t ws_size,
                              hipStream_t stream) {
  (void)in_sizes; (void)n_in; (void)out_size; (void)ws_size;
  const float* x    = (const float*)d_in[0];
  const int*   tpos = (const int*)d_in[1];
  const float* Wq   = (const float*)d_in[2];
  const float* Wk   = (const float*)d_in[3];
  const float* Wv   = (const float*)d_in[4];
  const float* Wo   = (const float*)d_in[5];
  float* out = (float*)d_out;

  const size_t nx = (size_t)Mrows * Dc;       // 16,777,216
  const size_t nw = (size_t)Dc * Dc;          //  4,194,304

  char* ws = (char*)d_ws;
  __bf16* Xb  = (__bf16*)ws;  ws += nx * sizeof(__bf16);
  __bf16* Wqb = (__bf16*)ws;  ws += nw * sizeof(__bf16);
  __bf16* Wkb = (__bf16*)ws;  ws += nw * sizeof(__bf16);
  __bf16* Wvb = (__bf16*)ws;  ws += nw * sizeof(__bf16);
  __bf16* Wob = (__bf16*)ws;  ws += nw * sizeof(__bf16);
  __bf16* Qhd = (__bf16*)ws;  ws += nx * sizeof(__bf16);
  __bf16* Khd = (__bf16*)ws;  ws += nx * sizeof(__bf16);
  __bf16* Vtd = (__bf16*)ws;  ws += nx * sizeof(__bf16);
  __bf16* Att = (__bf16*)ws;  ws += nx * sizeof(__bf16);

  // 1) convert inputs to bf16
  f32_to_bf16_kernel<<<(unsigned)((nx + 255) / 256), 256, 0, stream>>>(x, Xb, nx);
  f32_to_bf16_kernel<<<(unsigned)((nw + 255) / 256), 256, 0, stream>>>(Wq, Wqb, nw);
  f32_to_bf16_kernel<<<(unsigned)((nw + 255) / 256), 256, 0, stream>>>(Wk, Wkb, nw);
  f32_to_bf16_kernel<<<(unsigned)((nw + 255) / 256), 256, 0, stream>>>(Wv, Wvb, nw);
  f32_to_bf16_kernel<<<(unsigned)((nw + 255) / 256), 256, 0, stream>>>(Wo, Wob, nw);

  // 2) QKV projections (TDM-staged WMMA), epilogues scatter per-head layouts
  dim3 gblk(256);
  dim3 ggrd(Mrows / 64, Dc / 256);            // 128 x 8
  gemm_xwt<1><<<ggrd, gblk, 0, stream>>>(Xb, Wqb, Qhd, Mrows, Dc, Dc);
  gemm_xwt<1><<<ggrd, gblk, 0, stream>>>(Xb, Wkb, Khd, Mrows, Dc, Dc);
  gemm_xwt<2><<<ggrd, gblk, 0, stream>>>(Xb, Wvb, Vtd, Mrows, Dc, Dc);

  // 3) RoPE on Q, K (in place)
  const size_t nr = (size_t)Bc * Hc * Sc * (DKc / 2);   // 8,388,608
  rope_kernel<<<(unsigned)((nr + 255) / 256), 256, 0, stream>>>(Qhd, Khd, tpos, nr);

  // 4) causal flash attention (one wave per 16 query rows)
  attn_kernel<<<Bc * Hc * (Sc / 16), 32, 0, stream>>>(Qhd, Khd, Vtd, Att);

  // 5) output projection, f32 result
  gemm_xwt<0><<<ggrd, gblk, 0, stream>>>(Att, Wob, out, Mrows, Dc, Dc);
}
